// CompressedMultiHeadAttention_73675868995959
// MI455X (gfx1250) — compile-verified, matrix operand order checked by reading, not run on hardware
//
#include <hip/hip_runtime.h>
#include <hip/hip_bf16.h>

// ---------------------------------------------------------------------------
// CompressedMultiHeadAttention for MI455X (gfx1250, wave32, WMMA + TDM).
//
// Pipeline (dense bf16 WMMA GEMMs, fp32 accumulation, TDM tile staging):
//   K0  repack conv weights [o,c,cf] -> bf16 [o, cf*D + c];
//       convert w_qkv / w_out fp32 -> bf16 (so every GEMM B operand is bf16)
//   K1  qkv    = x @ w_qkv^T + b                  (A manual fp32, B TDM)
//   K2a kc     = conv-GEMM over qkv k-region      (A TDM, B TDM, K=4096)
//   K2b vcT    = conv-GEMM over v, stored TRANSPOSED [d][b*sc]
//   K3  scores = (q @ kc^T)/32, tril mask         (A TDM, B TDM, fp32 out)
//   K4  rowwise stats: max + sum(exp(s-max))
//   K5  attn   = exp(scores-max) @ vc / sum       (A manual fp32+exp, B TDM)
//   K6  out    = attn @ w_out^T + b_out           (A TDM, B TDM, fp32 out)
//
// GEMM core: 128x128 C-tile / 256 threads (8 waves = 4M x 2N), K chunks of 32
// staged as bf16 into ping-pong LDS buffers (row pitch 40 elems = 80 B so all
// fragment ds_load_b128 are 16-B aligned). TDM descriptors use pad_enable with
// pad_interval=3 (16 DWORDs) / pad_amount=3 (4 DWORDs) to produce that pitch.
// ---------------------------------------------------------------------------

typedef __attribute__((ext_vector_type(16))) __bf16        v16bf;
typedef __attribute__((ext_vector_type(8)))  float         v8f;
typedef __attribute__((ext_vector_type(4)))  unsigned int  u32x4;
typedef __attribute__((ext_vector_type(8)))  int           i32x8;
typedef __attribute__((ext_vector_type(4)))  int           i32x4;

union FragU { uint4 u[2]; v16bf v; };

#if defined(__HIP_DEVICE_COMPILE__) && defined(__has_builtin)
#  if __has_builtin(__builtin_amdgcn_tensor_load_to_lds) && \
      __has_builtin(__builtin_amdgcn_s_wait_tensorcnt)
#    define USE_TDM 1
#  endif
#endif
#ifndef USE_TDM
#  define USE_TDM 0
#endif

__device__ __forceinline__ unsigned short f2bf(float f) {
  unsigned u = __float_as_uint(f);
  u += 0x7fffu + ((u >> 16) & 1u);           // round-to-nearest-even
  return (unsigned short)(u >> 16);
}

#if USE_TDM
// Issue one TDM 2-D tile load: 128 rows x 32 bf16, row stride strideElems,
// into LDS at byte offset ldsByteOff (pitch 80 B via padding). Wave-level op.
// This toolchain exposes the 6-arg builtin:
//   (uint32x4 g0, int32x8 g1, int32x4 g2, int32x4 g3, int32x8 g4, i32 cpol)
__device__ __forceinline__ void tdm_load_tile(unsigned ldsByteOff,
                                              const unsigned short* gptr,
                                              long strideElems) {
  const unsigned long long ga = (unsigned long long)gptr;
  u32x4 g0;
  g0[0] = 1u;                                   // count=1 (valid user D#)
  g0[1] = ldsByteOff;                           // lds_addr
  g0[2] = (unsigned)(ga & 0xffffffffu);         // global_addr[31:0]
  g0[3] = (unsigned)((ga >> 32) & 0x1ffffffu)   // global_addr[56:32]
        | (2u << 30);                           // type = 2 ("image")
  const unsigned td0 = 0x7fffffffu;             // unbounded tensor dims
  const unsigned td1 = 0x7fffffffu;
  const unsigned long long s0 = (unsigned long long)strideElems;
  i32x8 g1;
  g1[0] = (1 << 16)                             // data_size = 2 bytes
        | (1 << 20)                             // pad_enable
        | (3 << 22)                             // pad_interval: 16 DWORDs
        | (3 << 25);                            // pad_amount:   4 DWORDs
  g1[1] = (int)((td0 & 0xffffu) << 16);                          // dim0 lo
  g1[2] = (int)(((td0 >> 16) & 0xffffu) | ((td1 & 0xffffu) << 16));
  g1[3] = (int)(((td1 >> 16) & 0xffffu) | (32u << 16));          // tile_dim0=32
  g1[4] = 128;                                  // tile_dim1=128, tile_dim2=0
  g1[5] = (int)(s0 & 0xffffffffu);              // tensor_dim0_stride lo
  g1[6] = (int)((s0 >> 32) & 0xffffu);          // stride hi, dim1_stride=0
  g1[7] = 0;
  const i32x4 z4 = {0, 0, 0, 0};                // 2-D tensor: groups 2/3 unused
  const i32x8 z8 = {0, 0, 0, 0, 0, 0, 0, 0};
  __builtin_amdgcn_tensor_load_to_lds(g0, g1, z4, z4, z8, 0);
}
#endif

// ---------------------------------------------------------------------------
// Generic 128x128-tile GEMM:  C[z] = epilogue( A[z] x B[z]^T )
//   A element (r, kk): Ap[ z*A_batch + r*A_rs + (kk>>10)*A_bs + (kk&1023) ]
//   B element (n, kk): Bp[ z*B_batch + n*B_ns + kk ]   (always bf16, [n][k])
//   A_MODE: 0 = fp32 (manual cvt stage), 1 = bf16 (TDM stage),
//           2 = fp32 + exp(s - rowmax) prologue (manual stage)
// ---------------------------------------------------------------------------
template <int A_MODE, bool MASK, bool HAS_BIAS, bool ROWSCALE, bool C_F32,
          bool C_TRANS>
__global__ __launch_bounds__(256) void gemm_wmma_bf16(
    const void* __restrict__ Ap, long A_rs, long A_bs, long A_batch,
    const unsigned short* __restrict__ Bp, long B_ns, long B_batch,
    const float* __restrict__ bias,
    const float* __restrict__ stats, long statsBatch,   // (max,sum) pairs
    void* __restrict__ Cp, long C_ld, long C_batch,
    int K, float scale, const int* __restrict__ maskFlag)
{
  // ping-pong buffers: A[buf] at elems buf*5120, B[buf] at 10240 + buf*5120
  __shared__ unsigned short lds[4 * 5120];      // 40 KB

  const int tid  = threadIdx.x;
  const int lane = tid & 31;
  const int wave = tid >> 5;
  const int wm   = wave & 3;                    // 4 waves along M (32 each)
  const int wn   = wave >> 2;                   // 2 waves along N (64 each)
  const int half = lane >> 4;
  const int lr   = lane & 15;
  const long z   = blockIdx.z;
  const long mBase = (long)blockIdx.y * 128;
  const long nBase = (long)blockIdx.x * 128;

  v8f acc[2][4];
  const v8f vzero = {0.f, 0.f, 0.f, 0.f, 0.f, 0.f, 0.f, 0.f};
  #pragma unroll
  for (int i = 0; i < 2; ++i)
    #pragma unroll
    for (int j = 0; j < 4; ++j) acc[i][j] = vzero;

  // ---- stage one 128x32 K-chunk (A and B tiles) into LDS buffer `buf` ----
  auto stage = [&](int k0, int buf) {
    const unsigned aOff = (unsigned)(buf * 5120) * 2u;            // bytes
    const unsigned bOff = (unsigned)(10240 + buf * 5120) * 2u;    // bytes
    const long bbase = z * B_batch + nBase * B_ns + k0;
    const long aRel  = ((long)(k0 >> 10)) * A_bs + (k0 & 1023);
#if USE_TDM
    if (wave == 0) {
      tdm_load_tile(bOff, Bp + bbase, B_ns);
      if (A_MODE == 1) {
        const long abase = z * A_batch + mBase * A_rs + aRel;
        tdm_load_tile(aOff, (const unsigned short*)Ap + abase, A_rs);
      }
    }
#else
    #pragma unroll
    for (int i = tid; i < 128 * 32; i += 256) {               // manual B
      const int n = i >> 5, cc = i & 31;
      lds[(bOff >> 1) + n * 40 + cc] = Bp[bbase + (long)n * B_ns + cc];
    }
    if (A_MODE == 1) {
      #pragma unroll
      for (int i = tid; i < 128 * 32; i += 256) {             // manual bf16 A
        const int r = i >> 5, cc = i & 31;
        const long off = z * A_batch + (mBase + r) * A_rs + aRel + cc;
        lds[(aOff >> 1) + r * 40 + cc] = ((const unsigned short*)Ap)[off];
      }
    }
#endif
    if (A_MODE != 1) {                                        // fp32 (+exp) A
      __builtin_prefetch((const char*)Ap +
                         (z * A_batch + (mBase + wave * 16) * A_rs + aRel) * 4,
                         0, 1);
      #pragma unroll
      for (int i = tid; i < 128 * 32; i += 256) {
        const int r = i >> 5, cc = i & 31;
        const long off = z * A_batch + (mBase + r) * A_rs + aRel + cc;
        float val = ((const float*)Ap)[off];
        if (A_MODE == 2) {
          const float mx = stats[(z * statsBatch + mBase + r) * 2];
          val = __expf(val - mx);            // unnormalized softmax prob
        }
        lds[(aOff >> 1) + r * 40 + cc] = f2bf(val);
      }
    }
  };

  const int nc = K / 32;
  stage(0, 0);
  for (int c = 0; c < nc; ++c) {
    const int buf = c & 1;
    if (c + 1 < nc) stage((c + 1) * 32, (c + 1) & 1);
#if USE_TDM
    // per-wave in-order TENSORcnt: with next chunk's loads outstanding,
    // waiting <= nTDM guarantees the current chunk has landed.
    if (c + 1 < nc) __builtin_amdgcn_s_wait_tensorcnt(A_MODE == 1 ? 2 : 1);
    else            __builtin_amdgcn_s_wait_tensorcnt(0);
#endif
    __syncthreads();

    const unsigned short* AsB = lds + buf * 5120;
    const unsigned short* BsB = lds + 10240 + buf * 5120;

    // fragment loads: CDNA5 16-bit 16x32 layout — lanes 0-15 hold
    // K{0..7,16..23}, lanes 16-31 hold K{8..15,24..31}; 2x ds_load_b128 each.
    FragU a[2], b[4];
    #pragma unroll
    for (int mt = 0; mt < 2; ++mt) {
      const unsigned short* p = AsB + (wm * 32 + mt * 16 + lr) * 40 + half * 8;
      a[mt].u[0] = *(const uint4*)(p);
      a[mt].u[1] = *(const uint4*)(p + 16);
    }
    #pragma unroll
    for (int nt = 0; nt < 4; ++nt) {
      const unsigned short* p = BsB + (wn * 64 + nt * 16 + lr) * 40 + half * 8;
      b[nt].u[0] = *(const uint4*)(p);
      b[nt].u[1] = *(const uint4*)(p + 16);
    }
    #pragma unroll
    for (int mt = 0; mt < 2; ++mt)
      #pragma unroll
      for (int nt = 0; nt < 4; ++nt)
        acc[mt][nt] = __builtin_amdgcn_wmma_f32_16x16x32_bf16(
            false, a[mt].v, false, b[nt].v, (short)0, acc[mt][nt], false, false);
    __syncthreads();   // protect buf before it is re-staged at c+2
  }

  // -------- epilogue: scale / bias / mask / rowscale / store --------
  const bool doMask = MASK ? (maskFlag != nullptr && *maskFlag != 0) : false;
  #pragma unroll
  for (int mt = 0; mt < 2; ++mt) {
    #pragma unroll
    for (int nt = 0; nt < 4; ++nt) {
      const long gn = nBase + wn * 64 + nt * 16 + lr;
      #pragma unroll
      for (int r = 0; r < 8; ++r) {
        const long gm = mBase + wm * 32 + mt * 16 + (r + half * 8);
        float v = acc[mt][nt][r] * scale;
        if (HAS_BIAS) v += bias[gn];
        if (MASK)     v = (!doMask || gn <= gm) ? v : -3.0e38f; // tril(S,Sc)
        if (ROWSCALE) v *= 1.0f / stats[(z * statsBatch + gm) * 2 + 1];
        const long co = C_TRANS ? (z * C_batch + gn * C_ld + gm)
                                : (z * C_batch + gm * C_ld + gn);
        if (C_F32) ((float*)Cp)[co] = v;
        else       ((unsigned short*)Cp)[co] = f2bf(v);
      }
    }
  }
}

// ---------------------------------------------------------------------------
// Conv weight repack: [o, c, cf] fp32 -> bf16 [o, cf*1024 + c]  (D=1024, CF=4)
// ---------------------------------------------------------------------------
__global__ __launch_bounds__(256) void repack_conv(
    const float* __restrict__ w, unsigned short* __restrict__ out)
{
  const long i  = (long)blockIdx.x * 256 + threadIdx.x;   // over 1024*1024*4
  const int  cf = (int)(i & 3);
  const long rc = i >> 2;
  const int  c  = (int)(rc & 1023);
  const long o  = rc >> 10;
  out[o * 4096 + (long)cf * 1024 + c] = f2bf(w[i]);
}

// Plain fp32 -> bf16 conversion (w_qkv, w_out one-time preconvert)
__global__ __launch_bounds__(256) void cvt_f32_bf16(
    const float* __restrict__ s, unsigned short* __restrict__ d)
{
  const long i = (long)blockIdx.x * 256 + threadIdx.x;
  d[i] = f2bf(s[i]);
}

// ---------------------------------------------------------------------------
// Rowwise softmax statistics over Sc=1024: stats[row] = {max, sum(exp(s-max))}
// One wave per row (8 rows / 256-thread block), shfl_xor tree reduce.
// ---------------------------------------------------------------------------
__global__ __launch_bounds__(256) void softmax_stats(
    const float* __restrict__ scores, float* __restrict__ stats)
{
  const int row  = blockIdx.x * 8 + (threadIdx.x >> 5);
  const int lane = threadIdx.x & 31;
  const float* p = scores + (long)row * 1024;

  float vals[32];
  float mx = -3.4e38f;
  #pragma unroll
  for (int i = 0; i < 32; ++i) { vals[i] = p[lane + i * 32]; mx = fmaxf(mx, vals[i]); }
  #pragma unroll
  for (int s = 16; s > 0; s >>= 1) mx = fmaxf(mx, __shfl_xor(mx, s, 32));
  float sum = 0.f;
  #pragma unroll
  for (int i = 0; i < 32; ++i) sum += __expf(vals[i] - mx);
  #pragma unroll
  for (int s = 16; s > 0; s >>= 1) sum += __shfl_xor(sum, s, 32);
  if (lane == 0) { stats[row * 2] = mx; stats[row * 2 + 1] = sum; }
}

// ---------------------------------------------------------------------------
extern "C" void kernel_launch(void* const* d_in, const int* in_sizes, int n_in,
                              void* d_out, int out_size, void* d_ws, size_t ws_size,
                              hipStream_t stream)
{
  (void)in_sizes; (void)n_in; (void)out_size; (void)ws_size;

  const float* x     = (const float*)d_in[0];   // [4,4096,1024]
  const float* w_qkv = (const float*)d_in[1];   // [3072,1024]
  const float* b_qkv = (const float*)d_in[2];   // [3072]
  const float* wk    = (const float*)d_in[3];   // [1024,1024,4]
  const float* bk    = (const float*)d_in[4];
  const float* wv    = (const float*)d_in[5];
  const float* bv    = (const float*)d_in[6];
  const float* w_out = (const float*)d_in[7];   // [1024,1024]
  const float* b_out = (const float*)d_in[8];
  const int*   mask  = (const int*)d_in[9];

  // ---- workspace layout (~243 MB) ----
  char* ws = (char*)d_ws;
  size_t off = 0;
  auto alloc = [&](size_t bytes) {
    off = (off + 255) & ~(size_t)255;
    void* p = ws + off;
    off += bytes;
    return p;
  };
  unsigned short* qkv     = (unsigned short*)alloc((size_t)16384 * 3072 * 2);
  unsigned short* wqkv_bf = (unsigned short*)alloc((size_t)3072 * 1024 * 2);
  unsigned short* wout_bf = (unsigned short*)alloc((size_t)1024 * 1024 * 2);
  unsigned short* wk_r    = (unsigned short*)alloc((size_t)1024 * 4096 * 2);
  unsigned short* wv_r    = (unsigned short*)alloc((size_t)1024 * 4096 * 2);
  unsigned short* kc      = (unsigned short*)alloc((size_t)4096 * 1024 * 2);
  unsigned short* vcT     = (unsigned short*)alloc((size_t)1024 * 4096 * 2);
  float*          scores  = (float*)         alloc((size_t)16384 * 1024 * 4);
  float*          stats   = (float*)         alloc((size_t)16384 * 2 * 4);
  unsigned short* attn    = (unsigned short*)alloc((size_t)16384 * 1024 * 2);

  // K0: weight preprocessing (one-time, all GEMM B operands become bf16)
  repack_conv<<<16384, 256, 0, stream>>>(wk, wk_r);
  repack_conv<<<16384, 256, 0, stream>>>(wv, wv_r);
  cvt_f32_bf16<<<12288, 256, 0, stream>>>(w_qkv, wqkv_bf);
  cvt_f32_bf16<<<4096, 256, 0, stream>>>(w_out, wout_bf);

  // K1: qkv = x @ w_qkv^T + b_qkv     (M=16384, N=3072, K=1024)
  gemm_wmma_bf16<0, false, true, false, false, false>
      <<<dim3(24, 128, 1), 256, 0, stream>>>(
          x, 1024, 0, 0,
          wqkv_bf, 1024, 0,
          b_qkv, nullptr, 0,
          qkv, 3072, 0,
          1024, 1.0f, nullptr);

  // K2a: kc = conv-compress(k)        (M=4096, N=1024, K=4096)
  //   A row r=(b*1024+sc): qkv[(4r+cf)*3072 + 1024 + c], kk = cf*1024+c
  gemm_wmma_bf16<1, false, true, false, false, false>
      <<<dim3(8, 32, 1), 256, 0, stream>>>(
          qkv + 1024, 12288, 3072, 0,
          wk_r, 4096, 0,
          bk, nullptr, 0,
          kc, 1024, 0,
          4096, 1.0f, nullptr);

  // K2b: vcT = conv-compress(v), stored transposed [d=1024][b*sc=4096]
  gemm_wmma_bf16<1, false, true, false, false, true>
      <<<dim3(8, 32, 1), 256, 0, stream>>>(
          qkv + 2048, 12288, 3072, 0,
          wv_r, 4096, 0,
          bv, nullptr, 0,
          vcT, 4096, 0,
          4096, 1.0f, nullptr);

  // K3: scores = (q @ kc^T)/32, tril mask  (batched z=4; M=4096,N=1024,K=1024)
  gemm_wmma_bf16<1, true, false, false, true, false>
      <<<dim3(8, 32, 4), 256, 0, stream>>>(
          qkv, 3072, 0, (long)4096 * 3072,
          kc, 1024, (long)1024 * 1024,
          nullptr, nullptr, 0,
          scores, 1024, (long)4096 * 1024,
          1024, 0.03125f, mask);

  // K4: rowwise max + sumexp over all 16384 rows
  softmax_stats<<<2048, 256, 0, stream>>>(scores, stats);

  // K5: attn = exp(scores-max) @ vc, /sum in epilogue
  //     B(n=d, kk=key) = vcT[d*4096 + z*1024 + kk]  (k-contiguous)
  gemm_wmma_bf16<2, false, false, true, false, false>
      <<<dim3(8, 32, 4), 256, 0, stream>>>(
          scores, 1024, 0, (long)4096 * 1024,
          vcT, 4096, 1024,
          nullptr, stats, 4096,
          attn, 1024, (long)4096 * 1024,
          1024, 1.0f, nullptr);

  // K6: out = attn @ w_out^T + b_out  (M=16384, N=1024, K=1024) -> fp32 d_out
  gemm_wmma_bf16<1, false, true, false, true, false>
      <<<dim3(8, 128, 1), 256, 0, stream>>>(
          attn, 1024, 0, 0,
          wout_bf, 1024, 0,
          b_out, nullptr, 0,
          d_out, 1024, 0,
          1024, 1.0f, nullptr);
}
